// Model_81183471829235
// MI455X (gfx1250) — compile-verified, matrix operand order checked by reading, not run on hardware
//
#include <hip/hip_runtime.h>

// ---------------------------------------------------------------------------
// Peephole LSTM for MI455X (gfx1250, wave32), bf16 WMMA + f32 accumulate.
//   1. convert x/Wx/Wh/Wfc f32 -> bf16, bias4h = bx + bh
//   2. xg[T,B,4H] = x @ Wx^T + bias4h   (WMMA GEMM, async-to-LDS double buffer)
//   3. persistent recurrence kernel, 512 steps, device-wide barrier/step,
//      Wh slice staged ONCE in LDS via async-to-LDS, cell state in registers,
//      h double-buffered bf16 in L2
//   4. out = hT @ Wfc^T + bfc           (same WMMA GEMM)
// ---------------------------------------------------------------------------

typedef __bf16 bf16;
typedef __attribute__((ext_vector_type(8)))  bf16  v8bf;
typedef __attribute__((ext_vector_type(16))) bf16  v16bf;
typedef __attribute__((ext_vector_type(8)))  float v8f;

static constexpr int kB  = 256;
static constexpr int kT  = 512;
static constexpr int kI  = 1024;
static constexpr int kH  = 1024;
static constexpr int kG4 = 4096;   // 4*H
static constexpr int kO  = 1024;
static constexpr int kRecWG = 64;  // persistent workgroups in recurrence

// ---------------- helpers ---------------------------------------------------

__device__ __forceinline__ unsigned short f2bf(float f) {
  unsigned u = __float_as_uint(f);
  u += 0x7FFFu + ((u >> 16) & 1u);          // round-to-nearest-even
  return (unsigned short)(u >> 16);
}

__device__ __forceinline__ float fsig(float x) {
  return 1.0f / (1.0f + __expf(-x));
}

__device__ __forceinline__ float ftanh_fast(float x) {
  float ax = fabsf(x);
  float e  = __expf(-2.0f * ax);
  float t  = (1.0f - e) / (1.0f + e);
  return x < 0.0f ? -t : t;
}

// Per-lane async copy of 16 bytes global -> LDS (ASYNCcnt-tracked).
// lds_off = addrspace(3) byte address (low 32 bits of the flat pointer).
__device__ __forceinline__ void async_copy16(unsigned lds_off, const void* g) {
  asm volatile("global_load_async_to_lds_b128 %0, %1, off"
               :: "v"(lds_off), "v"((unsigned long long)(size_t)g)
               : "memory");
}

__device__ __forceinline__ void wait_async0() {
  asm volatile("s_wait_asynccnt 0x0" ::: "memory");
}

// Load one 16x32 bf16 WMMA operand fragment for this lane (ISA 7.12.2):
// two contiguous 16B chunks at +0 and +16 elements.
__device__ __forceinline__ v16bf load_frag(const bf16* p) {
  v8bf lo = *reinterpret_cast<const v8bf*>(p);
  v8bf hi = *reinterpret_cast<const v8bf*>(p + 16);
  v16bf r;
#pragma unroll
  for (int e = 0; e < 8; ++e) { r[e] = lo[e]; r[e + 8] = hi[e]; }
  return r;
}

__device__ __forceinline__ v8f wmma_bf16(v16bf a, v16bf b, v8f c) {
  return __builtin_amdgcn_wmma_f32_16x16x32_bf16(
      false, a, false, b, (short)0, c, false, false);
}

// ---------------- stage 1: conversions --------------------------------------

__global__ void cvt_f32_bf16_kernel(const float* __restrict__ in,
                                    unsigned short* __restrict__ out,
                                    size_t n4) {
  size_t i      = (size_t)blockIdx.x * blockDim.x + threadIdx.x;
  size_t stride = (size_t)gridDim.x * blockDim.x;
  for (; i < n4; i += stride) {
    float4 v = reinterpret_cast<const float4*>(in)[i];
    ushort4 o;
    o.x = f2bf(v.x); o.y = f2bf(v.y); o.z = f2bf(v.z); o.w = f2bf(v.w);
    reinterpret_cast<ushort4*>(out)[i] = o;
  }
}

__global__ void bias_sum_kernel(const float* __restrict__ a,
                                const float* __restrict__ b,
                                float* __restrict__ o, int n) {
  int i = blockIdx.x * blockDim.x + threadIdx.x;
  if (i < n) o[i] = a[i] + b[i];
}

// ---------------- stages 2 & 4: bf16 WMMA GEMM with async LDS staging -------
// C[M,N] = A[M,K] * W[N,K]^T + bias[N]
// Block: 256 threads = 8 waves, tile 128(M) x 128(N), K-step 32.
// A/B tiles (8KB each) double-buffered in LDS, filled by
// global_load_async_to_lds_b128, consumed as ds_load_b128 fragments.
// mode 0: C[m*N + n]   |   mode 1: scatter to xg[t,b,4H] with m = b*kT + t

__global__ __launch_bounds__(256)
void gemm_bf16_wmma(const bf16* __restrict__ A, const bf16* __restrict__ W,
                    const float* __restrict__ bias, float* __restrict__ C,
                    int M, int N, int K, int mode) {
  __shared__ bf16 sA[2][128 * 32];   // 2 x 8KB
  __shared__ bf16 sB[2][128 * 32];   // 2 x 8KB

  const int tid   = threadIdx.x;
  const int lane  = tid & 31;
  const int wave  = tid >> 5;
  const int waveM = wave >> 1;
  const int waveN = wave & 1;
  const int rlo   = lane & 15;          // M (A) / N (B,C) within 16-tile
  const int khi   = (lane >> 4) << 3;   // K sub-offset for fragments
  const int mhi   = khi;                // M sub-offset for C fragment

  const int mblk = blockIdx.x * 128;
  const int nblk = blockIdx.y * 128;

  // staging: 512 chunks of 16B per matrix tile, 2 per thread
  const int r0 = tid >> 2,          q0 = (tid & 3) << 3;          // chunk tid
  const int r1 = (tid + 256) >> 2,  q1 = (tid & 3) << 3;          // chunk tid+256
  const unsigned baseA = (unsigned)(size_t)(&sA[0][0]);
  const unsigned baseB = (unsigned)(size_t)(&sB[0][0]);

  auto stage = [&](int bi, int k) {
    const unsigned bo = (unsigned)bi * (128 * 32 * 2);
    async_copy16(baseA + bo + (unsigned)(r0 * 64 + q0 * 2),
                 A + (size_t)(mblk + r0) * K + k + q0);
    async_copy16(baseA + bo + (unsigned)(r1 * 64 + q1 * 2),
                 A + (size_t)(mblk + r1) * K + k + q1);
    async_copy16(baseB + bo + (unsigned)(r0 * 64 + q0 * 2),
                 W + (size_t)(nblk + r0) * K + k + q0);
    async_copy16(baseB + bo + (unsigned)(r1 * 64 + q1 * 2),
                 W + (size_t)(nblk + r1) * K + k + q1);
  };

  v8f acc[2][4];
#pragma unroll
  for (int mt = 0; mt < 2; ++mt)
#pragma unroll
    for (int nt = 0; nt < 4; ++nt) acc[mt][nt] = (v8f)(0.0f);

  stage(0, 0);

  for (int k = 0; k < K; k += 32) {
    const int cur = (k >> 5) & 1;
    wait_async0();
    __syncthreads();
    if (k + 32 < K) stage(cur ^ 1, k + 32);

    v16bf a[2], b[4];
#pragma unroll
    for (int mt = 0; mt < 2; ++mt)
      a[mt] = load_frag(&sA[cur][(waveM * 32 + mt * 16 + rlo) * 32 + khi]);
#pragma unroll
    for (int nt = 0; nt < 4; ++nt)
      b[nt] = load_frag(&sB[cur][(waveN * 64 + nt * 16 + rlo) * 32 + khi]);
#pragma unroll
    for (int mt = 0; mt < 2; ++mt)
#pragma unroll
      for (int nt = 0; nt < 4; ++nt)
        acc[mt][nt] = wmma_bf16(a[mt], b[nt], acc[mt][nt]);

    __syncthreads();
  }

  const int m0 = mblk + waveM * 32;
  const int n0 = nblk + waveN * 64;
#pragma unroll
  for (int nt = 0; nt < 4; ++nt) {
    const int n  = n0 + nt * 16 + rlo;
    const float bv = bias[n];
#pragma unroll
    for (int mt = 0; mt < 2; ++mt) {
#pragma unroll
      for (int r = 0; r < 8; ++r) {
        const int m = m0 + mt * 16 + r + mhi;
        const float v = acc[mt][nt][r] + bv;
        size_t idx;
        if (mode == 1) {
          const int bi = m >> 9;          // m / kT
          const int ti = m & (kT - 1);    // m % kT
          idx = (((size_t)ti * kB + bi) << 12) + n;   // [t][b][4H]
        } else {
          idx = (size_t)m * N + n;
        }
        C[idx] = v;
      }
    }
  }
}

// ---------------- stage 3: persistent recurrence ----------------------------
// 64 WGs x 256 threads. WG owns hidden-unit slice [blockIdx.x*16, +16).
// Wave w owns batch rows [w*32, +32). The WG's 64 Wh rows (4 gates x 16
// units, 128KB bf16) are staged ONCE into LDS with async-to-LDS and reused
// for all 512 steps. Cell state lives in registers; h double-buffered bf16.

__global__ __launch_bounds__(256)
void lstm_recurrence(const float* __restrict__ xg,   // [T,B,4H] f32
                     const bf16* __restrict__ Wh,    // [4H,H] bf16
                     const float* __restrict__ c2c,  // [3H] f32
                     bf16* __restrict__ hbuf,        // [2,B,H] bf16
                     unsigned* __restrict__ ctr) {
  extern __shared__ bf16 sWh[];          // [64][1024] = 128KB

  const int lane = threadIdx.x & 31;
  const int wave = threadIdx.x >> 5;
  const int rlo  = lane & 15;
  const int khi  = (lane >> 4) << 3;
  const int mhi  = khi;

  const int jbase = blockIdx.x * 16;
  const int j     = jbase + rlo;
  const int mbase = wave * 32;

  // one-time Wh slice stage: 8192 chunks of 16B, 32 per thread
  {
    const unsigned sbase = (unsigned)(size_t)(&sWh[0]);
#pragma unroll 4
    for (int i = 0; i < 32; ++i) {
      const int c   = threadIdx.x + (i << 8);
      const int row = c >> 7;                 // 128 chunks per 2KB row
      const int col = (c & 127) << 3;         // element offset in row
      const bf16* g =
          Wh + (size_t)((row >> 4) * kH + jbase + (row & 15)) * kH + col;
      async_copy16(sbase + (unsigned)(row * 2048 + col * 2), g);
    }
  }

  const float ci = c2c[j];
  const float cf = c2c[kH + j];
  const float co = c2c[2 * kH + j];

  float creg[2][8];
#pragma unroll
  for (int mt = 0; mt < 2; ++mt)
#pragma unroll
    for (int r = 0; r < 8; ++r) creg[mt][r] = 0.0f;

  wait_async0();
  __syncthreads();

  for (int t = 0; t < kT; ++t) {
    const bf16* hs = hbuf + (size_t)(t & 1) * (kB * kH);
    bf16*       hd = hbuf + (size_t)((t + 1) & 1) * (kB * kH);

    v8f acc[2][4];
#pragma unroll
    for (int mt = 0; mt < 2; ++mt)
#pragma unroll
      for (int g = 0; g < 4; ++g) acc[mt][g] = (v8f)(0.0f);

#pragma unroll 2
    for (int k = 0; k < kH; k += 32) {
      v16bf a[2], b[4];
#pragma unroll
      for (int mt = 0; mt < 2; ++mt)
        a[mt] = load_frag(hs + (size_t)(mbase + mt * 16 + rlo) * kH + k + khi);
#pragma unroll
      for (int g = 0; g < 4; ++g)
        b[g] = load_frag(&sWh[(g * 16 + rlo) * kH + k + khi]);
#pragma unroll
      for (int mt = 0; mt < 2; ++mt)
#pragma unroll
        for (int g = 0; g < 4; ++g)
          acc[mt][g] = wmma_bf16(a[mt], b[g], acc[mt][g]);
    }

    const float* xgt = xg + (size_t)t * (kB * kG4);
#pragma unroll
    for (int mt = 0; mt < 2; ++mt) {
#pragma unroll
      for (int r = 0; r < 8; ++r) {
        const int m = mbase + mt * 16 + r + mhi;
        const float* xr = xgt + (size_t)m * kG4 + j;
        const float iv = acc[mt][0][r] + xr[0];
        const float fv = acc[mt][1][r] + xr[kH];
        const float gv = acc[mt][2][r] + xr[2 * kH];
        const float ov = acc[mt][3][r] + xr[3 * kH];

        const float c_old = creg[mt][r];
        const float ig = fsig(iv + ci * c_old);
        const float fg = fsig(fv + cf * c_old);
        const float cn = fg * c_old + ig * ftanh_fast(gv);
        const float og = fsig(ov + co * cn);
        const float hn = og * ftanh_fast(cn);
        creg[mt][r] = cn;
        reinterpret_cast<unsigned short*>(hd)[(size_t)m * kH + j] = f2bf(hn);

        // warm next step's xg lines (global_prefetch_b8)
        if (t + 1 < kT) __builtin_prefetch(xr + kB * kG4, 0, 0);
      }
    }

    // device-wide barrier: all h stores visible before next step's reads.
    __threadfence();
    __syncthreads();
    if (threadIdx.x == 0) {
      __hip_atomic_fetch_add(ctr, 1u, __ATOMIC_RELEASE, __HIP_MEMORY_SCOPE_AGENT);
      const unsigned target = (unsigned)(t + 1) * (unsigned)kRecWG;
      while (__hip_atomic_load(ctr, __ATOMIC_ACQUIRE, __HIP_MEMORY_SCOPE_AGENT) < target) {
        __builtin_amdgcn_s_sleep(2);
      }
    }
    __syncthreads();
    __threadfence();
  }
}

// ---------------- launch ----------------------------------------------------

extern "C" void kernel_launch(void* const* d_in, const int* in_sizes, int n_in,
                              void* d_out, int out_size, void* d_ws, size_t ws_size,
                              hipStream_t stream) {
  (void)in_sizes; (void)n_in; (void)out_size; (void)ws_size;

  const float* x   = (const float*)d_in[0];   // [B,T,I]
  const float* Wx  = (const float*)d_in[1];   // [4H,I]
  const float* bx  = (const float*)d_in[2];   // [4H]
  const float* Wh  = (const float*)d_in[3];   // [4H,H]
  const float* bh  = (const float*)d_in[4];   // [4H]
  const float* c2c = (const float*)d_in[5];   // [3H]
  const float* Wfc = (const float*)d_in[6];   // [O,H]
  const float* bfc = (const float*)d_in[7];   // [O]
  float* out = (float*)d_out;                 // [B,O]

  char* ws = (char*)d_ws;
  size_t off = 0;
  auto alloc = [&](size_t bytes) -> char* {
    char* p = ws + off;
    off += (bytes + 1023) & ~(size_t)1023;
    return p;
  };
  bf16*  x_bf    = (bf16*)alloc((size_t)kB * kT * kI * 2);    // 256 MB
  bf16*  Wx_bf   = (bf16*)alloc((size_t)kG4 * kI * 2);        //   8 MB
  bf16*  Wh_bf   = (bf16*)alloc((size_t)kG4 * kH * 2);        //   8 MB
  bf16*  Wfc_bf  = (bf16*)alloc((size_t)kO * kH * 2);         //   2 MB
  float* bias4h  = (float*)alloc((size_t)kG4 * 4);
  bf16*  hbuf    = (bf16*)alloc((size_t)2 * kB * kH * 2);     //   1 MB
  unsigned* ctr  = (unsigned*)alloc(256);
  float* xg      = (float*)alloc((size_t)kT * kB * kG4 * 4);  //   2 GB

  hipMemsetAsync(ctr, 0, 256, stream);
  hipMemsetAsync(hbuf, 0, (size_t)2 * kB * kH * 2, stream);

  // stage 1: conversions
  {
    size_t n4 = (size_t)kB * kT * kI / 4;
    cvt_f32_bf16_kernel<<<8192, 256, 0, stream>>>(x, (unsigned short*)x_bf, n4);
    n4 = (size_t)kG4 * kI / 4;
    cvt_f32_bf16_kernel<<<2048, 256, 0, stream>>>(Wx, (unsigned short*)Wx_bf, n4);
    n4 = (size_t)kG4 * kH / 4;
    cvt_f32_bf16_kernel<<<2048, 256, 0, stream>>>(Wh, (unsigned short*)Wh_bf, n4);
    n4 = (size_t)kO * kH / 4;
    cvt_f32_bf16_kernel<<<1024, 256, 0, stream>>>(Wfc, (unsigned short*)Wfc_bf, n4);
    bias_sum_kernel<<<kG4 / 256, 256, 0, stream>>>(bx, bh, bias4h, kG4);
  }

  // stage 2: xg[T,B,4H] = x @ Wx^T + (bx+bh)
  {
    dim3 grid((kB * kT) / 128, kG4 / 128);
    gemm_bf16_wmma<<<grid, 256, 0, stream>>>(x_bf, Wx_bf, bias4h, xg,
                                             kB * kT, kG4, kI, /*mode=*/1);
  }

  // stage 3: persistent recurrence (Wh slice in 128KB dynamic LDS)
  lstm_recurrence<<<kRecWG, 256, 131072, stream>>>(xg, Wh_bf, c2c, hbuf, ctr);

  // stage 4: out = hT @ Wfc^T + bfc ; hT lives in hbuf[0] after 512 steps
  {
    dim3 grid(kB / 128, kO / 128);
    gemm_bf16_wmma<<<grid, 256, 0, stream>>>(hbuf, Wfc_bf, bfc, out,
                                             kB, kO, kH, /*mode=*/0);
  }
}
